// GNNVerdictPredictor_82635170775089
// MI455X (gfx1250) — matrix operand order repaired and checked by reading, not compiled, not run on hardware
//
#include <hip/hip_runtime.h>
#include <hip/hip_bf16.h>

typedef __attribute__((ext_vector_type(16))) __bf16 v16bf;
typedef __attribute__((ext_vector_type(8)))  __bf16 v8bf;
typedef __attribute__((ext_vector_type(8)))  float  v8f;

#define HD 256   // hidden dim
#define DIN 768  // input dim

// ---------------------------------------------------------------- utilities
__global__ void fill_zero_kernel(float* __restrict__ p, long long n) {
    long long i = (long long)blockIdx.x * blockDim.x + threadIdx.x;
    if (i < n) p[i] = 0.0f;
}

__global__ void degree_kernel(const int* __restrict__ src, const int* __restrict__ dst,
                              float* __restrict__ deg_out, float* __restrict__ deg_in, int E) {
    int e = blockIdx.x * blockDim.x + threadIdx.x;
    if (e < E) {
        atomicAdd(&deg_out[src[e]], 1.0f);
        atomicAdd(&deg_in[dst[e]], 1.0f);
    }
}

// in-place: deg -> deg^-1/2 (clamped >= 1)
__global__ void norm_kernel(float* __restrict__ deg_out, float* __restrict__ deg_in, int n) {
    int i = blockIdx.x * blockDim.x + threadIdx.x;
    if (i < n) {
        deg_out[i] = rsqrtf(fmaxf(deg_out[i], 1.0f));
        deg_in[i]  = rsqrtf(fmaxf(deg_in[i], 1.0f));
    }
}

// agg[dst] += hs[src]; hs is bf16 (L2-resident), agg fp32.
// 32 threads per edge, 8 bf16 (16B) per thread -> 8 fp32 atomics.
__global__ void scatter_kernel(const int* __restrict__ src, const int* __restrict__ dst,
                               const __bf16* __restrict__ hs, float* __restrict__ agg, int E) {
    long long gid = (long long)blockIdx.x * blockDim.x + threadIdx.x;
    long long e = gid >> 5;
    int c = (int)(gid & 31);
    if (e >= E) return;
    int s = src[e], d = dst[e];
    v8bf v = *(const v8bf*)(hs + (size_t)s * HD + c * 8);
    float* p = agg + (size_t)d * HD + c * 8;
    #pragma unroll
    for (int i = 0; i < 8; ++i) atomicAdd(p + i, (float)v[i]);
}

// ---------------------------------------------------------------- WMMA GEMM
// Cout[M,Nc](bf16) = post( (A[M,K] * rs_in[row]) @ W[K,Nc] + bias[col] )
// post: optional relu, then optional * rs_out[row]. A,W fp32 row-major ->
// bf16 on the LDS staging path; fp32 accumulate via v_wmma_f32_16x16x32_bf16.
// Block: 128 thr = 4 wave32, tile 64(M) x 64(N), K-step 32.
// GUARD=false: caller guarantees all 64 rows of the tile are in range.
template <bool RS_IN, bool RS_OUT, bool RELU, bool GUARD>
__global__ __launch_bounds__(128) void gemm_bf16_wmma(
    const float* __restrict__ A, const float* __restrict__ W,
    const float* __restrict__ bias,
    const float* __restrict__ rs_in,
    const float* __restrict__ rs_out,
    __bf16* __restrict__ Cout,
    int M, int K, int Nc, int rowOffset)
{
    __shared__ __align__(32) __bf16 Alds[64][32];  // [row][k]
    __shared__ __align__(32) __bf16 Wlds[64][32];  // [n][k]  (K-transposed)

    const int tid  = threadIdx.x;
    const int wave = tid >> 5;
    const int lane = tid & 31;
    const int half = lane >> 4;
    const int lrow = lane & 15;

    const int rowBase = rowOffset + blockIdx.y * 64;
    const int colBase = blockIdx.x * 64;

    v8f zero8 = {0.f, 0.f, 0.f, 0.f, 0.f, 0.f, 0.f, 0.f};
    v8f acc[4];
    #pragma unroll
    for (int t = 0; t < 4; ++t) acc[t] = zero8;

    for (int k0 = 0; k0 < K; k0 += 32) {
        // stage A tile: 64 rows x 32 k (2048 f32), 4 float4 per thread
        #pragma unroll
        for (int i = 0; i < 4; ++i) {
            int q  = i * 128 + tid;      // 0..511
            int r  = q >> 3;             // row in tile 0..63
            int cc = (q & 7) << 2;       // k in tile 0,4,..,28
            int grow = rowBase + r;
            float4 v = make_float4(0.f, 0.f, 0.f, 0.f);
            bool ok = !GUARD || (grow < M);
            if (ok) v = *(const float4*)(A + (size_t)grow * K + k0 + cc);
            if constexpr (RS_IN) {
                float s = ok ? rs_in[grow] : 0.f;
                v.x *= s; v.y *= s; v.z *= s; v.w *= s;
            }
            Alds[r][cc + 0] = (__bf16)v.x;
            Alds[r][cc + 1] = (__bf16)v.y;
            Alds[r][cc + 2] = (__bf16)v.z;
            Alds[r][cc + 3] = (__bf16)v.w;
        }
        // stage W tile: 32 k x 64 n (2048 f32), stored transposed [n][k]
        #pragma unroll
        for (int i = 0; i < 4; ++i) {
            int q  = i * 128 + tid;      // 0..511
            int kr = q >> 4;             // k in tile 0..31
            int nc = (q & 15) << 2;      // n in tile 0,4,..,60
            float4 v = *(const float4*)(W + (size_t)(k0 + kr) * Nc + colBase + nc);
            Wlds[nc + 0][kr] = (__bf16)v.x;
            Wlds[nc + 1][kr] = (__bf16)v.y;
            Wlds[nc + 2][kr] = (__bf16)v.z;
            Wlds[nc + 3][kr] = (__bf16)v.w;
        }
        __syncthreads();

        // A fragment: rows [wave*16, wave*16+16); per-lane K octets
        // {half*8..half*8+7} and {16+half*8 .. 16+half*8+7}
        const int ar = wave * 16 + lrow;
        union { v16bf v; v8bf h[2]; } afr;
        afr.h[0] = *(const v8bf*)&Alds[ar][half * 8];
        afr.h[1] = *(const v8bf*)&Alds[ar][16 + half * 8];

        #pragma unroll
        for (int t = 0; t < 4; ++t) {
            // B fragment: col N = t*16 + lrow, K octets {half*8..half*8+7} and {16+half*8..16+half*8+7}
            const int bn = t * 16 + lrow;
            union { v16bf v; v8bf h[2]; } bu;
            bu.h[0] = *(const v8bf*)&Wlds[bn][half * 8];
            bu.h[1] = *(const v8bf*)&Wlds[bn][16 + half * 8];
            v16bf bfr = bu.v;
            acc[t] = __builtin_amdgcn_wmma_f32_16x16x32_bf16(
                false, afr.v, false, bfr, (short)0, acc[t], false, false);
        }
        __syncthreads();
    }

    // epilogue: VGPR i -> (M = i + 8*half, N = lrow) within each 16x16 tile
    const int m0 = rowBase + wave * 16;
    #pragma unroll
    for (int t = 0; t < 4; ++t) {
        const int ncol = colBase + t * 16 + lrow;
        const float bv = bias[ncol];
        #pragma unroll
        for (int i = 0; i < 8; ++i) {
            int row = m0 + i + half * 8;
            if (!GUARD || row < M) {
                float v = acc[t][i] + bv;
                if constexpr (RELU) v = fmaxf(v, 0.f);
                if constexpr (RS_OUT) v *= rs_out[row];
                Cout[(size_t)row * Nc + ncol] = (__bf16)v;
            }
        }
    }
}

// ---------------------------------------------------------------- head MLP
__global__ __launch_bounds__(128) void head_kernel(
    const __bf16* __restrict__ h, const int* __restrict__ claim_idx,
    const float* __restrict__ Wc1, const float* __restrict__ bc1,
    const float* __restrict__ Wc2, const float* __restrict__ bc2,
    float* __restrict__ out)
{
    __shared__ float claim[HD];
    __shared__ float hid[HD / 2];
    const int t = threadIdx.x;
    const int ci = claim_idx[0];
    claim[t]       = (float)h[(size_t)ci * HD + t];
    claim[t + 128] = (float)h[(size_t)ci * HD + t + 128];
    __syncthreads();
    float a = bc1[t];
    for (int k = 0; k < HD; ++k) a += claim[k] * Wc1[k * (HD / 2) + t];
    hid[t] = fmaxf(a, 0.f);
    __syncthreads();
    if (t < 3) {
        float v = bc2[t];
        for (int k = 0; k < HD / 2; ++k) v += hid[k] * Wc2[k * 3 + t];
        out[t] = v;
    }
}

// ---------------------------------------------------------------- launcher
template <bool RS_IN, bool RS_OUT, bool RELU>
static void launch_gemm(const float* A, const float* W, const float* bias,
                        const float* rs_in, const float* rs_out, __bf16* Cout,
                        int M, int K, int Nc, hipStream_t stream) {
    const int fullBlocks = M / 64;          // unguarded bulk
    const int tailRows   = M - fullBlocks * 64;
    if (fullBlocks > 0) {
        dim3 g(Nc / 64, fullBlocks);
        gemm_bf16_wmma<RS_IN, RS_OUT, RELU, false><<<g, 128, 0, stream>>>(
            A, W, bias, rs_in, rs_out, Cout, M, K, Nc, 0);
    }
    if (tailRows > 0) {
        dim3 g(Nc / 64, 1);
        gemm_bf16_wmma<RS_IN, RS_OUT, RELU, true><<<g, 128, 0, stream>>>(
            A, W, bias, rs_in, rs_out, Cout, M, K, Nc, fullBlocks * 64);
    }
}

extern "C" void kernel_launch(void* const* d_in, const int* in_sizes, int n_in,
                              void* d_out, int out_size, void* d_ws, size_t ws_size,
                              hipStream_t stream) {
    const float* X     = (const float*)d_in[0];
    const int*   esrc  = (const int*)d_in[1];
    const int*   edst  = (const int*)d_in[2];
    const int*   claim = (const int*)d_in[3];
    const float* W_in  = (const float*)d_in[4];
    const float* b_in  = (const float*)d_in[5];
    const float* W_gnn = (const float*)d_in[6];
    const float* b_gnn = (const float*)d_in[7];
    const float* Wc1   = (const float*)d_in[8];
    const float* bc1   = (const float*)d_in[9];
    const float* Wc2   = (const float*)d_in[10];
    const float* bc2   = (const float*)d_in[11];

    const int Nn = in_sizes[0] / DIN;   // 100000
    const int E  = in_sizes[1];         // 3200000

    float* ws      = (float*)d_ws;
    float*  deg_out = ws;                          // [Nn] -> norm_src
    float*  deg_in  = ws + Nn;                     // [Nn] -> norm_dst
    float*  agg     = ws + 2 * (size_t)Nn;         // [Nn, 256] fp32
    __bf16* hs      = (__bf16*)(agg + (size_t)Nn * HD);  // [Nn, 256] bf16

    // degrees -> symmetric norms
    {
        long long n = 2LL * Nn;
        fill_zero_kernel<<<(int)((n + 255) / 256), 256, 0, stream>>>(deg_out, n);
    }
    degree_kernel<<<(E + 255) / 256, 256, 0, stream>>>(esrc, edst, deg_out, deg_in, E);
    norm_kernel<<<(Nn + 255) / 256, 256, 0, stream>>>(deg_out, deg_in, Nn);

    // hs = (X @ W_in + b_in) * norm_src   (no relu on input projection)
    launch_gemm<false, true, false>(X, W_in, b_in, nullptr, deg_out, hs,
                                    Nn, DIN, HD, stream);

    for (int l = 0; l < 3; ++l) {
        long long n = (long long)Nn * HD;
        fill_zero_kernel<<<(int)((n + 255) / 256), 256, 0, stream>>>(agg, n);

        long long swork = (long long)E * 32;
        scatter_kernel<<<(int)((swork + 255) / 256), 256, 0, stream>>>(esrc, edst, hs, agg, E);

        const float* Wl = W_gnn + (size_t)l * HD * HD;
        const float* bl = b_gnn + (size_t)l * HD;
        if (l == 2) {
            // last layer: no norm_src fold (head consumes raw h)
            launch_gemm<true, false, true>(agg, Wl, bl, deg_in, nullptr, hs,
                                           Nn, HD, HD, stream);
        } else {
            launch_gemm<true, true, true>(agg, Wl, bl, deg_in, deg_out, hs,
                                          Nn, HD, HD, stream);
        }
    }

    head_kernel<<<1, 128, 0, stream>>>(hs, claim, Wc1, bc1, Wc2, bc2, (float*)d_out);
}
